// CCAStereo_9766755631568
// MI455X (gfx1250) — matrix operand order repaired and checked by reading, not compiled, not run on hardware
//
#include <hip/hip_runtime.h>
#include <hip/hip_bf16.h>

// ---------------------------------------------------------------------------
// CCAStereo pipeline for MI455X (gfx1250, wave32, WMMA + TDM).
//  * all convs -> implicit GEMM via v_wmma_f32_16x16x32_bf16
//  * weights pre-swizzled to bf16 A-fragment layout (vector loads)
//  * 1x1 convs stage activations into LDS with tensor_load_to_lds (TDM)
//  * templated kernel shapes/epilogues: no runtime div in hot paths
// ---------------------------------------------------------------------------

typedef __bf16 v16bf __attribute__((ext_vector_type(16)));
typedef float  v8f   __attribute__((ext_vector_type(8)));
typedef unsigned int u32x4 __attribute__((ext_vector_type(4)));
typedef int          i32x4 __attribute__((ext_vector_type(4)));
typedef int          i32x8 __attribute__((ext_vector_type(8)));

#define CC   32
#define DD   12
#define HH   64
#define WW   128
#define HWSZ (HH * WW)          // 8192
#define DHW  (DD * HH * WW)     // 98304
#define FF   256
#define MM   8
#define CMH  4
#define LRELU 0.01f

#if defined(__has_builtin)
#if __has_builtin(__builtin_amdgcn_tensor_load_to_lds) && \
    __has_builtin(__builtin_amdgcn_s_wait_tensorcnt)
#define HAVE_TDM 1
#endif
#endif
#ifndef HAVE_TDM
#define HAVE_TDM 0
#endif

// fp32 -> bf16 round-to-nearest-even (bit trick, toolchain-safe)
__device__ __forceinline__ __bf16 f2bf(float f) {
    unsigned u = __builtin_bit_cast(unsigned, f);
    u += 0x7FFFu + ((u >> 16) & 1u);
    unsigned short h = (unsigned short)(u >> 16);
    return __builtin_bit_cast(__bf16, h);
}

// A-fragment K index for element e, lane-half h (contiguous: k = 16*half + e)
__device__ __forceinline__ int a_kof(int e, int half) {
    return half * 16 + e;
}

// ---------------------------------------------------------------------------
// Repack fp32 weights [Cout, Ktot] into bf16 WMMA A fragments:
//   frag[((coTile*nChunks + kc)*32 + lane)*16 + e]
// ---------------------------------------------------------------------------
__global__ __launch_bounds__(256) void repack_weights(
    const float* __restrict__ Wt, __bf16* __restrict__ frag,
    int Cout, int Ktot)
{
    const int nChunks = Ktot >> 5;
    const int coTiles = Cout >> 4;
    const int total = coTiles * nChunks * 32;
    const int id = blockIdx.x * 256 + threadIdx.x;
    if (id >= total) return;

    const int lane = id & 31;
    const int rest = id >> 5;
    const int kc   = rest % nChunks;
    const int coT  = rest / nChunks;
    const int half = lane >> 4;
    const int l16  = lane & 15;
    const int co   = coT * 16 + l16;

    __bf16* dst = frag + (size_t)id * 16;
    const float* src = Wt + (size_t)co * Ktot + (kc << 5);
    #pragma unroll
    for (int e = 0; e < 16; ++e)
        dst[e] = f2bf(src[a_kof(e, half)]);
}

// ---------------------------------------------------------------------------
// Spatial implicit-GEMM conv (KHxKW compile-time, zero pad), BN+LeakyReLU.
// Grid: x = pixel-pair tiles / 8 waves, y = coTile  (no runtime div/mod).
// One wave -> 16 out-chan x 32 pixels (2 tiles, shared A fragment, 2 WMMAs).
// ---------------------------------------------------------------------------
template <int KH, int KW>
__global__ __launch_bounds__(256) void conv_wmma(
    const __bf16* __restrict__ frag, const float* __restrict__ X,
    float* __restrict__ Y,
    const float* __restrict__ scale, const float* __restrict__ bias,
    int Cin, int H, int W, int nImg, int pad)
{
    constexpr int KHKW = KH * KW;
    const int HW      = H * W;
    const int P       = nImg * HW;
    const int nChunks = (Cin * KHKW) >> 5;

    const int lane = threadIdx.x & 31;
    const int pixPair = blockIdx.x * 8 + (threadIdx.x >> 5);
    const int coTile  = blockIdx.y;
    const int half = lane >> 4;
    const int l16  = lane & 15;

    // two pixel columns per lane
    const int pix0 = pixPair * 32 + l16;
    const int pix1 = pix0 + 16;
    const int img0 = pix0 / HW, rem0 = pix0 - img0 * HW;
    const int py0 = rem0 / W, px0 = rem0 - py0 * W;
    const int img1 = pix1 / HW, rem1 = pix1 - img1 * HW;
    const int py1 = rem1 / W, px1 = rem1 - py1 * W;
    const float* xi0 = X + (size_t)img0 * HW;
    const float* xi1 = X + (size_t)img1 * HW;

    const v16bf* afrag =
        (const v16bf*)(frag + ((size_t)coTile * nChunks * 32 + lane) * 16);

    v8f acc0 = {0.f,0.f,0.f,0.f,0.f,0.f,0.f,0.f};
    v8f acc1 = {0.f,0.f,0.f,0.f,0.f,0.f,0.f,0.f};

    for (int kc = 0; kc < nChunks; ++kc) {
        const v16bf a = afrag[(size_t)kc * 32];
        if (kc + 1 < nChunks)
            __builtin_prefetch(&afrag[(size_t)(kc + 1) * 32], 0, 0);

        v16bf b0, b1;
        #pragma unroll
        for (int e = 0; e < 16; ++e) {
            const int k = (kc << 5) + half * 16 + e;
            const int ci = k / KHKW;            // compile-time divisor
            const int t  = k - ci * KHKW;
            const int kh = t / KW;
            const int kw = t - kh * KW;
            const size_t cbase = (size_t)ci * P;

            const int yy0 = py0 + kh - pad, xx0 = px0 + kw - pad;
            float v0 = 0.f;
            if (yy0 >= 0 && yy0 < H && xx0 >= 0 && xx0 < W)
                v0 = xi0[cbase + yy0 * W + xx0];
            b0[e] = f2bf(v0);

            const int yy1 = py1 + kh - pad, xx1 = px1 + kw - pad;
            float v1 = 0.f;
            if (yy1 >= 0 && yy1 < H && xx1 >= 0 && xx1 < W)
                v1 = xi1[cbase + yy1 * W + xx1];
            b1[e] = f2bf(v1);
        }
        acc0 = __builtin_amdgcn_wmma_f32_16x16x32_bf16(
            false, a, false, b0, (short)0, acc0, false, false);
        acc1 = __builtin_amdgcn_wmma_f32_16x16x32_bf16(
            false, a, false, b1, (short)0, acc1, false, false);
    }

    #pragma unroll
    for (int r = 0; r < 8; ++r) {
        const int co = coTile * 16 + half * 8 + r;
        const float s = scale[co], bi = bias[co];
        float v0 = acc0[r] * s + bi;
        v0 = (v0 > 0.f) ? v0 : v0 * LRELU;
        Y[(size_t)co * P + pix0] = v0;
        float v1 = acc1[r] * s + bi;
        v1 = (v1 > 0.f) ? v1 : v1 * LRELU;
        Y[(size_t)co * P + pix1] = v1;
    }
}

// ---------------------------------------------------------------------------
// 1x1 conv GEMM: out[Cout, P] = W[Cout, Cin] * X[Cin, P].
// Each block stages X[0:Cin][p0:p0+64] into LDS via the Tensor Data Mover
// (fallback: cooperative copy), then 8 waves compute 16x16 tiles with WMMA.
// MODE 0: +bias | 1: bn+lrelu | 2: sigmoid(+bias)*gate
// ---------------------------------------------------------------------------
template <int MODE>
__global__ __launch_bounds__(256) void gemm_wmma(
    const __bf16* __restrict__ frag, const float* __restrict__ X,
    float* __restrict__ Y,
    const float* __restrict__ scale, const float* __restrict__ bias,
    const float* __restrict__ gate,
    int Cin, int Cout, int P)
{
    extern __shared__ float lds_x[];            // [Cin][64] fp32
    const int nChunks = Cin >> 5;
    const int coTiles = Cout >> 4;
    const int blockPix = blockIdx.x * 64;
    const int tid = threadIdx.x;

#if HAVE_TDM
    if (tid < 32) {                             // one wave issues the DMA
        const unsigned lds_off =
            (unsigned)(unsigned long long)(uintptr_t)(void*)lds_x;
        const unsigned long long ga =
            (unsigned long long)(uintptr_t)(const void*)(X + blockPix);
        u32x4 g0;
        g0.x = 1u;                              // count=1, user descriptor
        g0.y = lds_off;                         // LDS byte address
        g0.z = (unsigned)(ga & 0xffffffffull);  // global_addr[31:0]
        g0.w = (unsigned)((ga >> 32) & 0x1ffffffull) | (2u << 30); // type=2
        i32x8 g1;
        g1[0] = (2 << 16);                                  // data_size=4B
        g1[1] = (int)(((unsigned)P & 0xffffu) << 16);       // tensor_dim0 lo16
        g1[2] = (int)((((unsigned)P >> 16) & 0xffffu) |
                      (((unsigned)Cin & 0xffffu) << 16));   // dim0 hi / dim1 lo
        g1[3] = (int)((((unsigned)Cin >> 16) & 0xffffu) |
                      (64u << 16));                         // dim1 hi / tile_dim0=64
        g1[4] = (int)((unsigned)Cin & 0xffffu);             // tile_dim1=Cin
        g1[5] = (int)(unsigned)P;                           // dim0_stride lo32
        g1[6] = 0;                                          // stride hi / dim1_stride
        g1[7] = 0;
        i32x4 g2 = {0, 0, 0, 0};
        i32x4 g3 = {0, 0, 0, 0};
#if defined(__clang_major__) && (__clang_major__ >= 23)
        i32x8 g4 = {0, 0, 0, 0, 0, 0, 0, 0};
        __builtin_amdgcn_tensor_load_to_lds(g0, g1, g2, g3, g4, 0);
#else
        __builtin_amdgcn_tensor_load_to_lds(g0, g1, g2, g3, 0);
#endif
        __builtin_amdgcn_s_wait_tensorcnt(0);
    }
#else
    for (int i = tid; i < Cin * 64; i += 256) {
        const int kr = i >> 6;
        const int cc = i & 63;
        lds_x[i] = X[(size_t)kr * P + blockPix + cc];
    }
#endif
    __syncthreads();

    const int lane = tid & 31;
    const int wave = tid >> 5;
    const int half = lane >> 4;
    const int l16  = lane & 15;
    const int pairs = coTiles * 4;              // 4 pixel tiles per block

    for (int pair = wave; pair < pairs; pair += 8) {
        const int tIn    = pair & 3;
        const int coTile = pair >> 2;
        const v16bf* afrag =
            (const v16bf*)(frag + ((size_t)coTile * nChunks * 32 + lane) * 16);

        v8f acc = {0.f,0.f,0.f,0.f,0.f,0.f,0.f,0.f};
        for (int kc = 0; kc < nChunks; ++kc) {
            const v16bf a = afrag[(size_t)kc * 32];
            v16bf b;
            #pragma unroll
            for (int e = 0; e < 16; ++e) {
                const int k = (kc << 5) + half * 16 + e;
                b[e] = f2bf(lds_x[k * 64 + tIn * 16 + l16]);
            }
            acc = __builtin_amdgcn_wmma_f32_16x16x32_bf16(
                false, a, false, b, (short)0, acc, false, false);
        }

        const int pix = blockPix + tIn * 16 + l16;
        #pragma unroll
        for (int r = 0; r < 8; ++r) {
            const int co = coTile * 16 + half * 8 + r;
            const size_t idx = (size_t)co * P + pix;
            float v = acc[r];
            if constexpr (MODE == 1) {
                v = v * scale[co] + bias[co];
                v = (v > 0.f) ? v : v * LRELU;
            } else if constexpr (MODE == 2) {
                v = v + bias[co];
                v = 1.0f / (1.0f + __expf(-v));
                v *= gate[idx];
            } else {
                v = v + bias[co];
            }
            Y[idx] = v;
        }
    }
}

// ---------------------------------------------------------------------------
// Fused window attention (per head m, voxel p): 25 logits over cm=4 channels,
// softmax, excite accumulate from v (broadcast over d). Raw excite -> E.
// ---------------------------------------------------------------------------
__global__ __launch_bounds__(256) void attn_excite(
    const float* __restrict__ qk, const float* __restrict__ v,
    float* __restrict__ E)
{
    const int p = blockIdx.x * 256 + threadIdx.x;
    if (p >= DHW) return;
    const int m = blockIdx.y;

    const int d = p / HWSZ;
    const int r = p - d * HWSZ;
    const int h = r / WW;
    const int w = r - h * WW;

    const float* qk0 = qk + (size_t)(m * CMH + 0) * DHW;
    const float* qk1 = qk + (size_t)(m * CMH + 1) * DHW;
    const float* qk2 = qk + (size_t)(m * CMH + 2) * DHW;
    const float* qk3 = qk + (size_t)(m * CMH + 3) * DHW;

    const float q0 = qk0[p], q1 = qk1[p], q2 = qk2[p], q3 = qk3[p];

    float logits[25];
    float mx = -3.4e38f;
    #pragma unroll
    for (int i = 0; i < 5; ++i) {
        #pragma unroll
        for (int j = 0; j < 5; ++j) {
            const int t  = i * 5 + j;
            const int hh = h + i - 2;
            const int ww = w + j - 2;
            float s = 0.0f;                   // zero-padded window -> logit 0
            if (hh >= 0 && hh < HH && ww >= 0 && ww < WW) {
                const int off = d * HWSZ + hh * WW + ww;
                s = q0 * qk0[off] + q1 * qk1[off] + q2 * qk2[off] + q3 * qk3[off];
            }
            logits[t] = s;
            mx = fmaxf(mx, s);
        }
    }
    float den = 0.0f;
    #pragma unroll
    for (int t = 0; t < 25; ++t) {
        const float e = __expf(logits[t] - mx);
        logits[t] = e;
        den += e;
    }
    const float inv = 1.0f / den;

    const float* v0 = v + (size_t)(m * CMH + 0) * HWSZ;
    const float* v1 = v + (size_t)(m * CMH + 1) * HWSZ;
    const float* v2 = v + (size_t)(m * CMH + 2) * HWSZ;
    const float* v3 = v + (size_t)(m * CMH + 3) * HWSZ;

    float e0 = 0.f, e1 = 0.f, e2 = 0.f, e3 = 0.f;
    #pragma unroll
    for (int i = 0; i < 5; ++i) {
        #pragma unroll
        for (int j = 0; j < 5; ++j) {
            const int t  = i * 5 + j;
            const int hh = h + i - 2;
            const int ww = w + j - 2;
            if (hh >= 0 && hh < HH && ww >= 0 && ww < WW) {
                const float wg  = logits[t] * inv;
                const int   off = hh * WW + ww;
                e0 += wg * v0[off];
                e1 += wg * v1[off];
                e2 += wg * v2[off];
                e3 += wg * v3[off];
            }
        }
    }
    E[(size_t)(m * CMH + 0) * DHW + p] = e0;
    E[(size_t)(m * CMH + 1) * DHW + p] = e1;
    E[(size_t)(m * CMH + 2) * DHW + p] = e2;
    E[(size_t)(m * CMH + 3) * DHW + p] = e3;
}

// ---------------------------------------------------------------------------
// Channel LayerNorm (C=32) in place on E; one thread per voxel.
// ---------------------------------------------------------------------------
__global__ __launch_bounds__(256) void ln_kernel(
    float* __restrict__ E,
    const float* __restrict__ g, const float* __restrict__ b)
{
    const int p = blockIdx.x * 256 + threadIdx.x;
    if (p >= DHW) return;

    float x[CC];
    float mu = 0.0f;
    #pragma unroll
    for (int c = 0; c < CC; ++c) {
        x[c] = E[(size_t)c * DHW + p];
        mu += x[c];
    }
    mu *= (1.0f / CC);
    float var = 0.0f;
    #pragma unroll
    for (int c = 0; c < CC; ++c) {
        const float dlt = x[c] - mu;
        var += dlt * dlt;
    }
    var *= (1.0f / CC);
    const float rs = rsqrtf(var + 1e-5f);
    #pragma unroll
    for (int c = 0; c < CC; ++c)
        E[(size_t)c * DHW + p] = (x[c] - mu) * rs * g[c] + b[c];
}

// ---------------------------------------------------------------------------
extern "C" void kernel_launch(void* const* d_in, const int* in_sizes, int n_in,
                              void* d_out, int out_size, void* d_ws, size_t ws_size,
                              hipStream_t stream) {
    (void)in_sizes; (void)n_in; (void)out_size; (void)ws_size;

    const float* Cost     = (const float*)d_in[0];
    const float* Fmap     = (const float*)d_in[1];
    const float* qk_w1    = (const float*)d_in[2];
    const float* qk_bn_s  = (const float*)d_in[3];
    const float* qk_bn_b  = (const float*)d_in[4];
    const float* qk_w2    = (const float*)d_in[5];
    const float* qk_b2    = (const float*)d_in[6];
    const float* v_w1     = (const float*)d_in[7];
    const float* v_bn1_s  = (const float*)d_in[8];
    const float* v_bn1_b  = (const float*)d_in[9];
    const float* v_w2     = (const float*)d_in[10];
    const float* v_bn2_s  = (const float*)d_in[11];
    const float* v_bn2_b  = (const float*)d_in[12];
    const float* v_w3     = (const float*)d_in[13];
    const float* v_b3     = (const float*)d_in[14];
    const float* ln_g     = (const float*)d_in[15];
    const float* ln_b     = (const float*)d_in[16];
    const float* att_w1   = (const float*)d_in[17];
    const float* att_bn_s = (const float*)d_in[18];
    const float* att_bn_b = (const float*)d_in[19];
    const float* att_w2   = (const float*)d_in[20];
    const float* att_b2   = (const float*)d_in[21];
    const float* agg_w1   = (const float*)d_in[22];
    const float* agg_bn_s = (const float*)d_in[23];
    const float* agg_bn_b = (const float*)d_in[24];
    const float* agg_w2   = (const float*)d_in[25];
    const float* agg_b2   = (const float*)d_in[26];

    float* out = (float*)d_out;

    // fp32 workspace buffers
    float* ws = (float*)d_ws;
    const size_t VOL = (size_t)CC * DHW;
    float* A  = ws;                            // qk conv1 out; later att-hidden
    float* Bq = ws + VOL;                      // qk final;    later agg-hidden
    float* E  = ws + 2 * VOL;                  // excite (+LN in place)
    float* G  = ws + 3 * VOL;                  // gated cost
    float* V1 = ws + 4 * VOL;
    float* V2 = V1 + (size_t)FF * HWSZ;
    float* Vv = V2 + (size_t)(FF / 2) * HWSZ;
    float* wsEnd = Vv + (size_t)CC * HWSZ;

    // bf16 weight-fragment region
    __bf16* fb    = (__bf16*)wsEnd;
    __bf16* fq1   = fb;                        // 32*800
    __bf16* fv1   = fq1  + 32 * 800;           // 256*2304
    __bf16* fqk2  = fv1  + 256 * 2304;         // 32*32
    __bf16* fv2   = fqk2 + 32 * 32;            // 128*256
    __bf16* fv3   = fv2  + 128 * 256;          // 32*128
    __bf16* fatt1 = fv3  + 32 * 128;
    __bf16* fatt2 = fatt1 + 32 * 32;
    __bf16* fagg1 = fatt2 + 32 * 32;
    __bf16* fagg2 = fagg1 + 32 * 32;

    // ---- weight repack (bf16 A fragments) ----
    struct RP { const float* w; __bf16* f; int cout; int ktot; };
    const RP rps[9] = {
        {qk_w1, fq1, 32, 800}, {v_w1, fv1, 256, 2304}, {qk_w2, fqk2, 32, 32},
        {v_w2, fv2, 128, 256}, {v_w3, fv3, 32, 128},   {att_w1, fatt1, 32, 32},
        {att_w2, fatt2, 32, 32}, {agg_w1, fagg1, 32, 32}, {agg_w2, fagg2, 32, 32},
    };
    for (int i = 0; i < 9; ++i) {
        const int total = (rps[i].cout / 16) * (rps[i].ktot / 32) * 32;
        repack_weights<<<(total + 255) / 256, 256, 0, stream>>>(
            rps[i].w, rps[i].f, rps[i].cout, rps[i].ktot);
    }

    // ---- pipeline ----
    // 1) qk conv1 (5x5 per D slice) + BN + LReLU   [grid: 3072/8 x 2 coTiles]
    conv_wmma<5, 5><<<dim3((DHW / 32) / 8, CC / 16), 256, 0, stream>>>(
        fq1, Cost, A, qk_bn_s, qk_bn_b, CC, HH, WW, DD, 2);
    // 2) qk conv2 (1x1x1) + bias
    gemm_wmma<0><<<DHW / 64, 256, CC * 64 * 4, stream>>>(
        fqk2, A, Bq, nullptr, qk_b2, nullptr, CC, CC, DHW);
    // 3) v conv1 (3x3, 256->256) + BN + LReLU      [grid: 256/8 x 16 coTiles]
    conv_wmma<3, 3><<<dim3((HWSZ / 32) / 8, FF / 16), 256, 0, stream>>>(
        fv1, Fmap, V1, v_bn1_s, v_bn1_b, FF, HH, WW, 1, 1);
    // 4) v conv2 (1x1, 256->128) + BN + LReLU
    gemm_wmma<1><<<HWSZ / 64, 256, FF * 64 * 4, stream>>>(
        fv2, V1, V2, v_bn2_s, v_bn2_b, nullptr, FF, FF / 2, HWSZ);
    // 5) v conv3 (1x1, 128->32) + bias
    gemm_wmma<0><<<HWSZ / 64, 256, (FF / 2) * 64 * 4, stream>>>(
        fv3, V2, Vv, nullptr, v_b3, nullptr, FF / 2, CC, HWSZ);
    // 6) window attention + softmax + excite
    {
        dim3 grid((DHW + 255) / 256, MM, 1);
        attn_excite<<<grid, 256, 0, stream>>>(Bq, Vv, E);
    }
    // 7) LayerNorm in place
    ln_kernel<<<(DHW + 255) / 256, 256, 0, stream>>>(E, ln_g, ln_b);
    // 8) att hidden (1x1x1) + BN + LReLU
    gemm_wmma<1><<<DHW / 64, 256, CC * 64 * 4, stream>>>(
        fatt1, E, A, att_bn_s, att_bn_b, nullptr, CC, CC, DHW);
    // 9) att out + sigmoid gate * Cost
    gemm_wmma<2><<<DHW / 64, 256, CC * 64 * 4, stream>>>(
        fatt2, A, G, nullptr, att_b2, Cost, CC, CC, DHW);
    // 10) agg hidden (1x1x1) + BN + LReLU
    gemm_wmma<1><<<DHW / 64, 256, CC * 64 * 4, stream>>>(
        fagg1, G, Bq, agg_bn_s, agg_bn_b, nullptr, CC, CC, DHW);
    // 11) agg out (1x1x1) + bias -> d_out
    gemm_wmma<0><<<DHW / 64, 256, CC * 64 * 4, stream>>>(
        fagg2, Bq, out, nullptr, agg_b2, nullptr, CC, CC, DHW);
}
